// GInitResN_66108136620575
// MI455X (gfx1250) — matrix operand order repaired and checked by reading, not compiled, not run on hardware
//
#include <hip/hip_runtime.h>
#include <hip/hip_fp16.h>

typedef __attribute__((ext_vector_type(16))) _Float16 v16h;
typedef __attribute__((ext_vector_type(8)))  float    v8f;

#define D 128
#define C 5

// ---------- setup kernels ----------

// h = x (initial features), deg buffer zeroed
__global__ void k_init(const float* __restrict__ x, float* __restrict__ h,
                       float* __restrict__ deg, int N) {
  int i = blockIdx.x * blockDim.x + threadIdx.x;
  if (i < N * D) h[i] = x[i];
  if (i < N) deg[i] = 0.f;
}

// deg[dst] += 1
__global__ void k_deg(const int* __restrict__ dst, float* __restrict__ deg, int E) {
  int e = blockIdx.x * blockDim.x + threadIdx.x;
  if (e < E) atomicAdd(&deg[dst[e]], 1.0f);
}

// deg -> rsqrt(deg) in place
__global__ void k_dinv(float* __restrict__ deg, int N) {
  int i = blockIdx.x * blockDim.x + threadIdx.x;
  if (i < N) {
    float d = deg[i];
    deg[i] = d > 0.f ? rsqrtf(d) : 0.f;
  }
}

// ---------- per-layer kernels ----------

// one wave (32 lanes) per node: al[i]=h[i].att_l, ar[i]=h[i].att_r
__global__ void k_dots(const float* __restrict__ h, const float* __restrict__ attl,
                       const float* __restrict__ attr, float* __restrict__ al,
                       float* __restrict__ ar, int N) {
  int wave = (blockIdx.x * blockDim.x + threadIdx.x) >> 5;
  int lane = threadIdx.x & 31;
  if (wave >= N) return;
  const float* hr = h + (size_t)wave * D;
  float sl = 0.f, sr = 0.f;
#pragma unroll
  for (int j = 0; j < 4; ++j) {
    float v = hr[lane + 32 * j];
    sl += v * attl[lane + 32 * j];
    sr += v * attr[lane + 32 * j];
  }
#pragma unroll
  for (int m = 16; m >= 1; m >>= 1) {
    sl += __shfl_xor(sl, m, 32);
    sr += __shfl_xor(sr, m, 32);
  }
  if (lane == 0) { al[wave] = sl; ar[wave] = sr; }
}

// hn = eps*h0 = x ; zero stats
__global__ void k_initagg(const float* __restrict__ x, float* __restrict__ hn,
                          float* __restrict__ ssum, float* __restrict__ ssq, int N) {
  int i = blockIdx.x * blockDim.x + threadIdx.x;
  if (i < N * D) hn[i] = x[i];
  if (i < D) { ssum[i] = 0.f; ssq[i] = 0.f; }
}

// one wave per edge: hn[dst] += h[src] * tanh(al[src]+ar[dst]) * dinv[src]*dinv[dst]
__global__ void k_edge(const int* __restrict__ src, const int* __restrict__ dst,
                       const float* __restrict__ al, const float* __restrict__ ar,
                       const float* __restrict__ dinv, const float* __restrict__ h,
                       float* __restrict__ hn, int E) {
  int e = (blockIdx.x * blockDim.x + threadIdx.x) >> 5;
  int lane = threadIdx.x & 31;
  if (e >= E) return;
  int s = src[e], d = dst[e];
  float w = tanhf(al[s] + ar[d]) * dinv[s] * dinv[d];
  const float4 v = *(const float4*)(h + (size_t)s * D + lane * 4);
  float* o = hn + (size_t)d * D + lane * 4;
  atomicAdd(o + 0, v.x * w);
  atomicAdd(o + 1, v.y * w);
  atomicAdd(o + 2, v.z * w);
  atomicAdd(o + 3, v.w * w);
}

// h = elu(hn); accumulate per-feature sum & sumsq (each thread owns one feature column slice)
__global__ void k_stats(const float* __restrict__ hn, float* __restrict__ h,
                        float* __restrict__ ssum, float* __restrict__ ssq, int N) {
  int g = blockIdx.x * blockDim.x + threadIdx.x;
  int f = g & (D - 1);
  int row0 = g >> 7;
  int rstride = (gridDim.x * blockDim.x) >> 7;
  float s = 0.f, q = 0.f;
  for (int i = row0; i < N; i += rstride) {
    float v = hn[(size_t)i * D + f];
    v = v > 0.f ? v : expm1f(v);
    h[(size_t)i * D + f] = v;
    s += v;
    q += v * v;
  }
  atomicAdd(&ssum[f], s);
  atomicAdd(&ssq[f], q);
}

// fold GraphNorm into per-feature affine: h' = a*h + b
// E[(h - s*m)^2] = E[h^2] - (2s - s^2) m^2
__global__ void k_params(const float* __restrict__ ssum, const float* __restrict__ ssq,
                         const float* __restrict__ gw, const float* __restrict__ gb,
                         const float* __restrict__ gs, float* __restrict__ aff, int N) {
  int f = threadIdx.x;
  float invN = 1.0f / (float)N;
  float m = ssum[f] * invN;
  float msq = ssq[f] * invN;
  float s = gs[f];
  float var = msq - (2.f * s - s * s) * m * m;
  float istd = rsqrtf(var + 1e-5f);
  float w = gw[f];
  aff[f] = w * istd;
  aff[D + f] = gb[f] - w * s * m * istd;
}

__global__ void k_apply(float* __restrict__ h, const float* __restrict__ aff, int N) {
  int i = blockIdx.x * blockDim.x + threadIdx.x;
  if (i < N * D) {
    int f = i & (D - 1);
    h[i] = aff[f] * h[i] + aff[D + f];
  }
}

// ---------- final projection: WMMA f32 <- f16 x f16, one wave per 16-row tile ----------
__global__ void k_final(const float* __restrict__ h, const float* __restrict__ lw,
                        const float* __restrict__ lb, float* __restrict__ out, int N) {
  int tile = (blockIdx.x * blockDim.x + threadIdx.x) >> 5;
  int lane = threadIdx.x & 31;
  int ntiles = (N + 15) >> 4;
  if (tile >= ntiles) return;  // wave-uniform: EXEC stays all-1s for WMMA

  int half = lane >> 4;   // 0: lanes 0-15, 1: lanes 16-31
  int m = lane & 15;      // A row within tile / B,D column

  // B fragments (constant across tiles): lanes 0-15 hold K=0..15 of column m,
  // lanes 16-31 hold K=16..31 (contiguous k order, same as A). Columns >= C are zero pad.
  v16h bfrag[4];
#pragma unroll
  for (int kc = 0; kc < 4; ++kc) {
    int kbb = kc * 32 + half * 16;
#pragma unroll
    for (int j = 0; j < 16; ++j) {
      float bw = (m < C) ? lw[(kbb + j) * C + m] : 0.f;
      bfrag[kc][j] = (_Float16)bw;
    }
  }

  int row = tile * 16 + m;
  if (row >= N) row = N - 1;  // clamp (N=50000 is an exact multiple of 16)
  const float* hrow = h + (size_t)row * D;

  v8f acc = {};
#pragma unroll
  for (int kc = 0; kc < 4; ++kc) {
    // A layout: lanes 0-15 -> K 0..15, lanes 16-31 -> K 16..31 of the 32-chunk
    // (contiguous, matching B).
    int ka = kc * 32 + half * 16;
    v16h a;
#pragma unroll
    for (int j = 0; j < 8; ++j) {
      a[j]     = (_Float16)hrow[ka + j];
      a[j + 8] = (_Float16)hrow[ka + 8 + j];
    }
    acc = __builtin_amdgcn_wmma_f32_16x16x32_f16(false, a, false, bfrag[kc],
                                                 (short)0, acc, false, false);
  }

  // D layout: lane group writes rows r + 8*half, column m
  if (m < C) {
    float bias = lb[m];
#pragma unroll
    for (int r = 0; r < 8; ++r) {
      int orow = tile * 16 + r + 8 * half;
      if (orow < N) out[(size_t)orow * C + m] = acc[r] + bias;
    }
  }
}

// ---------- host ----------
extern "C" void kernel_launch(void* const* d_in, const int* in_sizes, int n_in,
                              void* d_out, int out_size, void* d_ws, size_t ws_size,
                              hipStream_t stream) {
  const float* x     = (const float*)d_in[0];
  const int*   ei    = (const int*)d_in[1];
  const float* att_l = (const float*)d_in[2];
  const float* att_r = (const float*)d_in[3];
  const float* gw    = (const float*)d_in[4];
  const float* gb    = (const float*)d_in[5];
  const float* gs    = (const float*)d_in[6];
  const float* lw    = (const float*)d_in[7];
  const float* lb    = (const float*)d_in[8];
  float* out = (float*)d_out;

  const int N = in_sizes[0] / D;   // 50000
  const int E = in_sizes[1] / 2;   // 600000
  const int L = in_sizes[2] / D;   // 4
  const int ND = N * D;

  float* ws   = (float*)d_ws;
  float* h    = ws;                       // N*D
  float* hn   = h + (size_t)ND;           // N*D
  float* dinv = hn + (size_t)ND;          // N
  float* al   = dinv + N;                 // N
  float* ar   = al + N;                   // N
  float* ssum = ar + N;                   // D
  float* ssq  = ssum + D;                 // D
  float* aff  = ssq + D;                  // 2*D

  const int* src = ei;
  const int* dst = ei + E;

  dim3 b(256);
  k_init<<<(ND + 255) / 256, b, 0, stream>>>(x, h, dinv, N);
  k_deg<<<(E + 255) / 256, b, 0, stream>>>(dst, dinv, E);
  k_dinv<<<(N + 255) / 256, b, 0, stream>>>(dinv, N);

  for (int l = 0; l < L; ++l) {
    k_dots<<<(N * 32 + 255) / 256, b, 0, stream>>>(h, att_l + l * D, att_r + l * D,
                                                   al, ar, N);
    k_initagg<<<(ND + 255) / 256, b, 0, stream>>>(x, hn, ssum, ssq, N);
    int eblocks = (int)(((long long)E * 32 + 255) / 256);
    k_edge<<<eblocks, b, 0, stream>>>(src, dst, al, ar, dinv, h, hn, E);
    k_stats<<<256, b, 0, stream>>>(hn, h, ssum, ssq, N);
    k_params<<<1, D, 0, stream>>>(ssum, ssq, gw + l * D, gb + l * D, gs + l * D,
                                  aff, N);
    k_apply<<<(ND + 255) / 256, b, 0, stream>>>(h, aff, N);
  }

  int ntiles = (N + 15) / 16;
  int fblocks = (ntiles * 32 + 255) / 256;
  k_final<<<fblocks, b, 0, stream>>>(h, lw, lb, out, N);
}
